// Fusion_68771016344082
// MI455X (gfx1250) — matrix operand order repaired and checked by reading, not compiled, not run on hardware
//
#include <hip/hip_runtime.h>
#include <math.h>

// ---------------------------------------------------------------- constants
enum : int {
  BB = 32, CC = 512, HH = 32, WW = 32, NN = 1024,   // N = H*W
  LL = 32, DD = 768, OO = 512, NHH = 8, HD = 64, VV = 30522
};
#define EPSN 1e-5f

typedef __attribute__((ext_vector_type(16))) _Float16 v16h;
typedef __attribute__((ext_vector_type(8)))  float    v8f;

// ---------------------------------------------------------------- WMMA utils
__device__ __forceinline__ v8f wmma_f16(v16h a, v16h b, v8f c) {
  return __builtin_amdgcn_wmma_f32_16x16x32_f16(
      false, a, false, b, (short)0, c, false, false);
}

// A fragment: row-major A tile 16(M) x 32(K), lda in elements.
__device__ __forceinline__ v16h load_frag_a(const _Float16* __restrict__ A,
                                            int lda, int lane) {
  const int m  = lane & 15;
  const int hi = (lane >> 4) << 3;
  const _Float16* row = A + (size_t)m * lda;
  v16h a;
#pragma unroll
  for (int i = 0; i < 8; ++i) {
    const int kb = ((i < 4) ? (2 * i) : (2 * i + 8)) + hi;
    a[2 * i]     = row[kb];
    a[2 * i + 1] = row[kb + 1];
  }
  return a;
}
__device__ __forceinline__ void load_a4(v16h (&a)[4], const _Float16* __restrict__ A0,
                                        int lda, int koff, int lane) {
#pragma unroll
  for (int mt = 0; mt < 4; ++mt)
    a[mt] = load_frag_a(A0 + (size_t)(mt * 16) * lda + koff, lda, lane);
}

// ---- swizzled B storage: each 32(K)x16(N) tile = 32 lanes x 16 halves (1KB).
__device__ __forceinline__ size_t bsw_idx(int r, int c, int KT) {
  const int kt = r >> 5, kin = r & 31, nt = c >> 4, nn = c & 15;
  const int lane = (((kin >> 3) & 1) << 4) + nn, j = (kin & 7) + ((kin >> 4) << 3);
  return ((((size_t)nt * KT + kt) << 5) + lane) * 16 + j;
}
__device__ __forceinline__ v16h load_frag_b_sw(const _Float16* __restrict__ base,
                                               int tile, int lane) {
  return *(const v16h*)(base + (((size_t)tile) << 9) + (lane << 4));
}

// ---- swizzled A position within one 16x32 tile (for LDS att matrix)
__device__ __forceinline__ int asw_idx(int m, int k) {
  const int hi = (k >> 3) & 1;
  const int kb = k - (hi << 3);
  const int i  = (kb < 8) ? (kb >> 1) : (((kb - 16) >> 1) + 4);
  const int j  = 2 * i + (k & 1);
  return ((m & 15) + (hi << 4)) * 16 + j;
}

// ---- 64(M) x 16(N) GEMM core: B ping-pong (K unrolled x2), A single-buffered.
// Live fragments: b0,b1,a[4] = 48 VGPRs -> no spills. KT must be even.
__device__ __forceinline__ void gemm64x16(const _Float16* __restrict__ A0, int lda,
                                          const _Float16* __restrict__ Bb, int tile0,
                                          int KT, int lane, v8f (&acc)[4]) {
  v16h b0 = load_frag_b_sw(Bb, tile0, lane);
  for (int kt = 0; kt < KT; kt += 2) {
    v16h b1 = load_frag_b_sw(Bb, tile0 + kt + 1, lane);
    v16h a[4];
    load_a4(a, A0, lda, kt * 32, lane);
#pragma unroll
    for (int mt = 0; mt < 4; ++mt) acc[mt] = wmma_f16(a[mt], b0, acc[mt]);
    if (kt + 2 < KT) {
      b0 = load_frag_b_sw(Bb, tile0 + kt + 2, lane);
      __builtin_prefetch(Bb + (((size_t)(tile0 + kt + 3)) << 9) + (lane << 4), 0, 3);
    }
    load_a4(a, A0, lda, (kt + 1) * 32, lane);
#pragma unroll
    for (int mt = 0; mt < 4; ++mt) acc[mt] = wmma_f16(a[mt], b1, acc[mt]);
  }
}

// ---------------------------------------------------------------- pack kernels
__global__ void k_cvt(const float* __restrict__ s, _Float16* __restrict__ d, int n) {
  for (int i = blockIdx.x * blockDim.x + threadIdx.x; i < n;
       i += gridDim.x * blockDim.x)
    d[i] = (_Float16)s[i];
}

// x [B][C][N] row-major f32 -> swizzled f16 B tiles per batch
__global__ void k_packX(const float* __restrict__ s, _Float16* __restrict__ d) {
  const size_t msz = (size_t)CC * NN;
  for (size_t i = (size_t)blockIdx.x * blockDim.x + threadIdx.x; i < msz * BB;
       i += (size_t)gridDim.x * blockDim.x) {
    const int b   = (int)(i >> 19);
    const int rem = (int)(i & (msz - 1));
    const int r = rem >> 10, c = rem & (NN - 1);
    d[(size_t)b * msz + bsw_idx(r, c, CC >> 5)] = (_Float16)s[i];
  }
}

// G [P x O] row-major -> swizzled B = G^T
__global__ void k_packB_tr(const float* __restrict__ s, _Float16* __restrict__ d) {
  for (int i = blockIdx.x * blockDim.x + threadIdx.x; i < OO * OO;
       i += gridDim.x * blockDim.x) {
    const int p = i >> 9, oin = i & (OO - 1);
    d[bsw_idx(oin, p, OO >> 5)] = (_Float16)s[i];
  }
}

// lfT[b] = gathered emb^T [D x L], stored swizzled
__global__ void k_gather(const float* __restrict__ emb, const int* __restrict__ lidx,
                         _Float16* __restrict__ d) {
  for (int i = blockIdx.x * blockDim.x + threadIdx.x; i < BB * DD * LL;
       i += gridDim.x * blockDim.x) {
    const int l = i & (LL - 1), t = i >> 5;
    const int dd_ = t % DD, b = t / DD;
    d[(size_t)b * DD * LL + bsw_idx(dd_, l, DD >> 5)] =
        (_Float16)emb[(size_t)lidx[b * LL + l] * DD + dd_];
  }
}

// ---------------------------------------------------------------- q = Wq @ x + bq
__global__ void k_qlin(const _Float16* __restrict__ Aw, const _Float16* __restrict__ Bsw,
                       const float* __restrict__ bias, _Float16* __restrict__ Cl) {
  const int lane = threadIdx.x & 31;
  const int gw   = (blockIdx.x * blockDim.x + threadIdx.x) >> 5;
  const int nt = gw & 63, og = (gw >> 6) & 7, b = gw >> 9;
  const int KT = CC >> 5;
  v8f acc[4] = {};
  gemm64x16(Aw + (size_t)(og * 64) * CC, CC,
            Bsw + (size_t)b * CC * NN, nt * KT, KT, lane, acc);
  const int col = lane & 15, rb = (lane >> 4) << 3;
#pragma unroll
  for (int mt = 0; mt < 4; ++mt)
#pragma unroll
    for (int r = 0; r < 8; ++r) {
      const int o = og * 64 + mt * 16 + rb + r, n = nt * 16 + col;
      Cl[((size_t)b * OO + o) * NN + n] = (_Float16)(acc[mt][r] + bias[o]);
    }
}

// InstanceNorm over N per (b,o); scatter into q16[B][NH][N][64] (linear A)
__global__ void k_norm_q(const _Float16* __restrict__ lin, _Float16* __restrict__ q16) {
  const int row = blockIdx.x;
  const int b = row >> 9, o = row & 511;
  const _Float16* p = lin + (size_t)row * NN;
  __shared__ float s1[256], s2[256];
  float s = 0.f, ss = 0.f;
  for (int i = threadIdx.x; i < NN; i += 256) { float x = (float)p[i]; s += x; ss += x * x; }
  s1[threadIdx.x] = s; s2[threadIdx.x] = ss; __syncthreads();
  for (int st = 128; st > 0; st >>= 1) {
    if (threadIdx.x < st) { s1[threadIdx.x] += s1[threadIdx.x + st];
                            s2[threadIdx.x] += s2[threadIdx.x + st]; }
    __syncthreads();
  }
  const float m = s1[0] * (1.f / NN);
  const float r = rsqrtf(s2[0] * (1.f / NN) - m * m + EPSN);
  const int h = o >> 6, dd = o & 63;
  for (int i = threadIdx.x; i < NN; i += 256)
    q16[(((size_t)(b * NHH + h)) * NN + i) * HD + dd] = (_Float16)(((float)p[i] - m) * r);
}

// ---------------------------------------------------------------- k,v convs
__global__ void k_kv(const _Float16* __restrict__ Wk16, const _Float16* __restrict__ Wv16,
                     const _Float16* __restrict__ lfTsw, const float* __restrict__ bk,
                     const float* __restrict__ bv, const float* __restrict__ l_mask,
                     _Float16* __restrict__ k16sw, _Float16* __restrict__ v16Tsw) {
  const int lane = threadIdx.x & 31;
  const int gw   = (blockIdx.x * blockDim.x + threadIdx.x) >> 5;
  const int lt = gw & 1, og = (gw >> 1) & 7, b = gw >> 4;
  const _Float16* Bb = lfTsw + (size_t)b * DD * LL;
  const int KT = DD >> 5;
  v8f ak[4] = {}, av[4] = {};
  gemm64x16(Wk16 + (size_t)(og * 64) * DD, DD, Bb, lt * KT, KT, lane, ak);
  gemm64x16(Wv16 + (size_t)(og * 64) * DD, DD, Bb, lt * KT, KT, lane, av);
  const int col = lane & 15, rb = (lane >> 4) << 3;
#pragma unroll
  for (int mt = 0; mt < 4; ++mt)
#pragma unroll
    for (int r = 0; r < 8; ++r) {
      const int o = og * 64 + mt * 16 + rb + r, l = lt * 16 + col;
      const float msk = l_mask[b * LL + l];
      const int h = o >> 6, dd = o & 63;
      k16sw[(size_t)(b * NHH + h) * HD * LL + bsw_idx(dd, l, 2)] =
          (_Float16)((ak[mt][r] + bk[o]) * msk);
      v16Tsw[(size_t)(b * NHH + h) * LL * HD + bsw_idx(l, dd, 1)] =
          (_Float16)((av[mt][r] + bv[o]) * msk);
    }
}

// ---------------------------------------------------------------- fused attention
__global__ void k_attn(const _Float16* __restrict__ q16, const _Float16* __restrict__ k16sw,
                       const _Float16* __restrict__ v16Tsw, const float* __restrict__ l_mask,
                       _Float16* __restrict__ outsw) {
  __shared__ float sims[8][16][LL];
  __shared__ __align__(32) _Float16 atts[8][512];
  const int wv = threadIdx.x >> 5, lane = threadIdx.x & 31;
  const int gw = blockIdx.x * 8 + wv;
  const int nt = gw & 63, h = (gw >> 6) & 7, b = gw >> 9;
  const _Float16* Aq = q16 + ((size_t)(b * NHH + h) * NN + nt * 16) * HD;
  const _Float16* Bk = k16sw + (size_t)(b * NHH + h) * HD * LL;
  v8f aL = {}, aH = {};
#pragma unroll
  for (int kt = 0; kt < 2; ++kt) {
    v16h af = load_frag_a(Aq + kt * 32, HD, lane);
    aL = wmma_f16(af, load_frag_b_sw(Bk, 0 * 2 + kt, lane), aL);
    aH = wmma_f16(af, load_frag_b_sw(Bk, 1 * 2 + kt, lane), aH);
  }
  const float scale = 0.044194173824159216f;    // 512^-0.5
  const int col = lane & 15, rb = (lane >> 4) << 3;
  const float bias0 = 10000.f * (l_mask[b * LL + col]      - 1.f);
  const float bias1 = 10000.f * (l_mask[b * LL + col + 16] - 1.f);
#pragma unroll
  for (int r = 0; r < 8; ++r) {
    const int m = rb + r;
    sims[wv][m][col]      = aL[r] * scale + bias0;
    sims[wv][m][col + 16] = aH[r] * scale + bias1;
  }
  __syncthreads();
  if (lane < 16) {
    float mx = -1e30f;
#pragma unroll
    for (int l = 0; l < LL; ++l) mx = fmaxf(mx, sims[wv][lane][l]);
    float e[LL], sum = 0.f;
#pragma unroll
    for (int l = 0; l < LL; ++l) { e[l] = __expf(sims[wv][lane][l] - mx); sum += e[l]; }
    const float inv = 1.f / sum;
#pragma unroll
    for (int l = 0; l < LL; ++l)
      atts[wv][asw_idx(lane, l)] = (_Float16)(e[l] * inv);
  }
  __syncthreads();
  v16h af = *(const v16h*)&atts[wv][lane << 4];
  const _Float16* Bv = v16Tsw + (size_t)(b * NHH + h) * LL * HD;
  _Float16* ob = outsw + (size_t)b * OO * NN;
#pragma unroll
  for (int dt = 0; dt < 4; ++dt) {
    v8f oc = {};
    oc = wmma_f16(af, load_frag_b_sw(Bv, dt, lane), oc);
#pragma unroll
    for (int r = 0; r < 8; ++r)
      ob[bsw_idx(h * HD + dt * 16 + col, nt * 16 + rb + r, OO >> 5)] = (_Float16)oc[r];
  }
}

// ---------------------------------------------------------------- w = Ww @ out + bw
__global__ void k_wconv(const _Float16* __restrict__ Aw, const _Float16* __restrict__ Bsw,
                        const float* __restrict__ bias, _Float16* __restrict__ Cl) {
  const int lane = threadIdx.x & 31;
  const int gw   = (blockIdx.x * blockDim.x + threadIdx.x) >> 5;
  const int nt = gw & 63, og = (gw >> 6) & 7, b = gw >> 9;
  const int KT = OO >> 5;
  v8f acc[4] = {};
  gemm64x16(Aw + (size_t)(og * 64) * OO, OO,
            Bsw + (size_t)b * OO * NN, nt * KT, KT, lane, acc);
  const int col = lane & 15, rb = (lane >> 4) << 3;
#pragma unroll
  for (int mt = 0; mt < 4; ++mt)
#pragma unroll
    for (int r = 0; r < 8; ++r) {
      const int o = og * 64 + mt * 16 + rb + r, n = nt * 16 + col;
      Cl[((size_t)b * OO + o) * NN + n] = (_Float16)(acc[mt][r] + bias[o]);
    }
}

// InstanceNorm over N per (b,o); write l_fea16[B][N][O] (linear)
__global__ void k_norm_w(const _Float16* __restrict__ lin, _Float16* __restrict__ lfea16) {
  const int row = blockIdx.x;
  const int b = row >> 9, o = row & 511;
  const _Float16* p = lin + (size_t)row * NN;
  __shared__ float s1[256], s2[256];
  float s = 0.f, ss = 0.f;
  for (int i = threadIdx.x; i < NN; i += 256) { float x = (float)p[i]; s += x; ss += x * x; }
  s1[threadIdx.x] = s; s2[threadIdx.x] = ss; __syncthreads();
  for (int st = 128; st > 0; st >>= 1) {
    if (threadIdx.x < st) { s1[threadIdx.x] += s1[threadIdx.x + st];
                            s2[threadIdx.x] += s2[threadIdx.x + st]; }
    __syncthreads();
  }
  const float m = s1[0] * (1.f / NN);
  const float r = rsqrtf(s2[0] * (1.f / NN) - m * m + EPSN);
  for (int i = threadIdx.x; i < NN; i += 256)
    lfea16[((size_t)b * NN + i) * OO + o] = (_Float16)(((float)p[i] - m) * r);
}

// ---------------------------------------------------------------- gate MLP
__global__ void k_gate1(const _Float16* __restrict__ lfea16, const _Float16* __restrict__ G1Tsw,
                        _Float16* __restrict__ hid) {
  const int lane = threadIdx.x & 31;
  const int gw   = (blockIdx.x * blockDim.x + threadIdx.x) >> 5;
  const int pt = gw & 31, rg = gw >> 5;
  const int KT = OO >> 5;
  v8f acc[4] = {};
  gemm64x16(lfea16 + (size_t)(rg * 64) * OO, OO, G1Tsw, pt * KT, KT, lane, acc);
  const int col = lane & 15, rb = (lane >> 4) << 3;
#pragma unroll
  for (int mt = 0; mt < 4; ++mt)
#pragma unroll
    for (int r = 0; r < 8; ++r)
      hid[((size_t)(rg * 64 + mt * 16 + rb + r)) * OO + pt * 16 + col] =
          (_Float16)fmaxf(acc[mt][r], 0.f);
}

__global__ void k_gate2(const _Float16* __restrict__ hid, const _Float16* __restrict__ G2Tsw,
                        const _Float16* __restrict__ lfea16, const float* __restrict__ x,
                        float* __restrict__ out) {
  const int lane = threadIdx.x & 31;
  const int gw   = (blockIdx.x * blockDim.x + threadIdx.x) >> 5;
  const int pt = gw & 31, rg = gw >> 5;
  const int KT = OO >> 5;
  v8f acc[4] = {};
  gemm64x16(hid + (size_t)(rg * 64) * OO, OO, G2Tsw, pt * KT, KT, lane, acc);
  const int col = lane & 15, rb = (lane >> 4) << 3;
#pragma unroll
  for (int mt = 0; mt < 4; ++mt)
#pragma unroll
    for (int r = 0; r < 8; ++r) {
      const size_t idx = ((size_t)(rg * 64 + mt * 16 + rb + r)) * OO + pt * 16 + col;
      out[idx] = x[idx] + tanhf(acc[mt][r]) * (float)lfea16[idx];
    }
}

// ---------------------------------------------------------------- launcher
extern "C" void kernel_launch(void* const* d_in, const int* in_sizes, int n_in,
                              void* d_out, int out_size, void* d_ws, size_t ws_size,
                              hipStream_t stream) {
  const float* x      = (const float*)d_in[0];
  const int*   l      = (const int*)  d_in[1];
  const float* l_mask = (const float*)d_in[2];
  const float* emb    = (const float*)d_in[3];
  const float* Wk     = (const float*)d_in[4];
  const float* bk     = (const float*)d_in[5];
  const float* Wv     = (const float*)d_in[6];
  const float* bv     = (const float*)d_in[7];
  const float* Wq     = (const float*)d_in[8];
  const float* bq     = (const float*)d_in[9];
  const float* Ww     = (const float*)d_in[10];
  const float* bw     = (const float*)d_in[11];
  const float* G1     = (const float*)d_in[12];
  const float* G2     = (const float*)d_in[13];
  float* out = (float*)d_out;

  char* p = (char*)d_ws;
  auto allocH = [&](size_t n) { _Float16* r = (_Float16*)p;
                                p += (n * sizeof(_Float16) + 255) & ~(size_t)255; return r; };
  _Float16* Wq16  = allocH((size_t)OO * CC);       // linear A
  _Float16* Wk16  = allocH((size_t)OO * DD);       // linear A
  _Float16* Wv16  = allocH((size_t)OO * DD);       // linear A
  _Float16* Ww16  = allocH((size_t)OO * OO);       // linear A
  _Float16* G1Tsw = allocH((size_t)OO * OO);       // swizzled B
  _Float16* G2Tsw = allocH((size_t)OO * OO);       // swizzled B
  _Float16* lfTsw = allocH((size_t)BB * DD * LL);  // swizzled B
  _Float16* k16sw = allocH((size_t)BB * OO * LL);  // swizzled B per (b,h)
  _Float16* v16Tsw= allocH((size_t)BB * OO * LL);  // swizzled B per (b,h)
  _Float16* bufA  = allocH((size_t)BB * CC * NN);  // x16 swizzled B, later lfea16 linear
  _Float16* bufB  = allocH((size_t)BB * OO * NN);  // q16 linear, later hid16 linear
  _Float16* outsw = allocH((size_t)BB * OO * NN);  // attn out, swizzled B
  _Float16* linb  = allocH((size_t)BB * OO * NN);  // qlin, later wlin (f16)
  (void)in_sizes; (void)n_in; (void)out_size; (void)ws_size;

  const dim3 T(256);
  k_cvt<<<2048, T, 0, stream>>>(Wq, Wq16, OO * CC);
  k_cvt<<<2048, T, 0, stream>>>(Wk, Wk16, OO * DD);
  k_cvt<<<2048, T, 0, stream>>>(Wv, Wv16, OO * DD);
  k_cvt<<<2048, T, 0, stream>>>(Ww, Ww16, OO * OO);
  k_packB_tr<<<1024, T, 0, stream>>>(G1, G1Tsw);
  k_packB_tr<<<1024, T, 0, stream>>>(G2, G2Tsw);
  k_packX<<<4096, T, 0, stream>>>(x, bufA);
  k_gather<<<1024, T, 0, stream>>>(emb, l, lfTsw);

  k_qlin<<<2048, T, 0, stream>>>(Wq16, bufA, bq, linb);
  k_norm_q<<<BB * OO, T, 0, stream>>>(linb, bufB);

  k_kv<<<64, T, 0, stream>>>(Wk16, Wv16, lfTsw, bk, bv, l_mask, k16sw, v16Tsw);

  k_attn<<<2048, T, 0, stream>>>(bufB, k16sw, v16Tsw, l_mask, outsw);

  k_wconv<<<2048, T, 0, stream>>>(Ww16, outsw, bw, linb);
  k_norm_w<<<BB * OO, T, 0, stream>>>(linb, bufA);

  k_gate1<<<2048, T, 0, stream>>>(bufA, G1Tsw, bufB);
  k_gate2<<<2048, T, 0, stream>>>(bufB, G2Tsw, bufA, x, out);
}
